// MultiHeadAttention_15341623182037
// MI455X (gfx1250) — matrix operand order repaired and checked by reading, not compiled, not run on hardware
//
#include <hip/hip_runtime.h>
#include <hip/hip_bf16.h>

// MHA: B=4, N=2048, D_IN=D_OUT=1024, H=16, HD=64.  f16 WMMA + TDM staging.

typedef _Float16 half_t;
typedef __attribute__((ext_vector_type(16))) _Float16 v16h;
typedef __attribute__((ext_vector_type(8)))  _Float16 v8h;
typedef __attribute__((ext_vector_type(8)))  float    v8f;
typedef __attribute__((ext_vector_type(4)))  unsigned int u32x4;
typedef __attribute__((ext_vector_type(8)))  int      i32x8;
typedef __attribute__((ext_vector_type(4)))  int      i32x4;

#define DEV __device__ __forceinline__

constexpr int BATCH = 4;
constexpr int NSEQ  = 2048;
constexpr int DMODEL = 1024;
constexpr int NHEAD = 16;
constexpr int HDIM  = 64;
constexpr int MROWS = BATCH * NSEQ;  // 8192
constexpr int KDIM  = DMODEL;        // 1024
constexpr int KSTEPS = KDIM / 64;    // 16

#if __has_builtin(__builtin_amdgcn_tensor_load_to_lds) && \
    __has_builtin(__builtin_amdgcn_s_wait_tensorcnt)
#define USE_TDM 1
#else
#define USE_TDM 0
#endif

DEV v8f vzero() { v8f z = {0.f,0.f,0.f,0.f,0.f,0.f,0.f,0.f}; return z; }

DEV v16h make_frag(v8h lo, v8h hi) {
  v16h r;
#pragma unroll
  for (int i = 0; i < 8; ++i) { r[i] = lo[i]; r[8 + i] = hi[i]; }
  return r;
}

DEV v8f wmma_f16(v16h a, v16h b, v8f c) {
  return __builtin_amdgcn_wmma_f32_16x16x32_f16(false, a, false, b, (short)0, c,
                                                false, false);
}

// Generic LDS pointer -> LDS byte offset (aperture: addr[31:0] = LDS offset).
DEV unsigned int lds_addr_of(const void* p) {
  return (unsigned int)(uintptr_t)p;
}

#if USE_TDM
// TDM 2D tile load: [tile_rows x 64] f16 tile of a [total_rows x 1024] f16
// row-major matrix starting at `g`, into LDS at lds_off with 16B padding per
// 128B row (-> LDS row stride 144B = 72 halves).
DEV void tdm_load_tile64(unsigned int lds_off, const half_t* g,
                         unsigned int total_rows, unsigned int tile_rows) {
  unsigned long long ga = (unsigned long long)(uintptr_t)g;
  u32x4 g0;
  g0[0] = 1u;                                   // count=1, user mode
  g0[1] = lds_off;                              // lds_addr
  g0[2] = (unsigned int)ga;                     // global_addr[31:0]
  g0[3] = (unsigned int)(ga >> 32) | (2u << 30);// global_addr[56:32] | type=2
  i32x8 g1;
  g1[0] = (1 << 16)      // data_size = 2 bytes
        | (1 << 20)      // pad_enable
        | (4 << 22)      // pad_interval: 32 DWORDs (128B)
        | (3 << 25);     // pad_amount: 4 DWORDs (16B)
  g1[1] = (int)(1024u << 16);                   // tensor_dim0 = 1024 (lo16)
  g1[2] = (int)((total_rows & 0xFFFFu) << 16);  // dim0 hi=0 | tensor_dim1 lo16
  g1[3] = (int)(64u << 16);                     // dim1 hi=0 | tile_dim0 = 64
  g1[4] = (int)tile_rows;                       // tile_dim1 | tile_dim2=0
  g1[5] = 1024;                                 // tensor_dim0_stride lo32
  g1[6] = 0;                                    // stride hi | dim1_stride lo
  g1[7] = 0;
  i32x4 z4 = {0, 0, 0, 0};
#if defined(__clang_major__) && __clang_major__ >= 23
  i32x8 z8 = {0, 0, 0, 0, 0, 0, 0, 0};
  __builtin_amdgcn_tensor_load_to_lds(g0, g1, z4, z4, z8, 0);
#else
  __builtin_amdgcn_tensor_load_to_lds(g0, g1, z4, z4, 0);
#endif
}
#endif

// One k-step of the 128x64 block tile: 8 wmma per wave (2 mt x 2 nt x 2 kk).
DEV void gemm_compute_step(const half_t (*Asb)[72], const half_t (*Bsb)[72],
                           int lane, int wm, int wn, v8f acc[2][2]) {
  const int cl = lane & 15;
  const int hi = lane >> 4;
#pragma unroll
  for (int kk = 0; kk < 2; ++kk) {
    v16h af[2], bf[2];
#pragma unroll
    for (int mt = 0; mt < 2; ++mt) {
      const half_t* p = &Asb[wm + mt * 16 + cl][kk * 32 + (hi << 3)];
      af[mt] = make_frag(*(const v8h*)p, *(const v8h*)(p + 16));
    }
#pragma unroll
    for (int nt = 0; nt < 2; ++nt) {
      const half_t* p = &Bsb[wn + nt * 16 + cl][kk * 32 + (hi << 3)];
      bf[nt] = make_frag(*(const v8h*)p, *(const v8h*)(p + 16));
    }
#pragma unroll
    for (int mt = 0; mt < 2; ++mt)
#pragma unroll
      for (int nt = 0; nt < 2; ++nt)
        acc[mt][nt] = wmma_f16(af[mt], bf[nt], acc[mt][nt]);
  }
}

// Main loop: C(128x64 block) = A[m0:m0+128,:] . Bt[n0:n0+64,:]^T
// As: [2*128][72] LDS rows (double buffer), Bs: [2*64][72].
DEV void gemm_mainloop(const half_t* __restrict__ A, const half_t* __restrict__ Bt,
                       unsigned int rowsA, unsigned int rowsB, int m0, int n0,
                       int tid, half_t (*As)[72], half_t (*Bs)[72],
                       v8f acc[2][2]) {
  const int lane = tid & 31, wid = tid >> 5;
  const int wm = (wid & 3) * 32;
  const int wn = (wid >> 2) * 32;
#if USE_TDM
  const half_t* Ap = A + (size_t)m0 * KDIM;
  const half_t* Bp = Bt + (size_t)n0 * KDIM;
  if (wid == 0) {
    tdm_load_tile64(lds_addr_of(&As[0][0]), Ap, rowsA, 128);
    tdm_load_tile64(lds_addr_of(&Bs[0][0]), Bp, rowsB, 64);
  }
  for (int s = 0; s < KSTEPS; ++s) {
    const int cur = s & 1;
    __syncthreads();  // previous users of buf[(s+1)&1] are done
    if (wid == 0) {
      if (s + 1 < KSTEPS) {
        tdm_load_tile64(lds_addr_of(&As[(1 - cur) * 128][0]),
                        Ap + (s + 1) * 64, rowsA, 128);
        tdm_load_tile64(lds_addr_of(&Bs[(1 - cur) * 64][0]),
                        Bp + (s + 1) * 64, rowsB, 64);
        __builtin_amdgcn_s_wait_tensorcnt(2);  // pair(s) done, pair(s+1) inflight
      } else {
        __builtin_amdgcn_s_wait_tensorcnt(0);
      }
    }
    __syncthreads();  // buf[cur] ready for everyone
    gemm_compute_step((const half_t(*)[72]) & As[cur * 128][0],
                      (const half_t(*)[72]) & Bs[cur * 64][0], lane, wm, wn, acc);
  }
#else
  // Fallback: manual global->VGPR->LDS staging, single buffer.
  for (int k0 = 0; k0 < KDIM; k0 += 64) {
#pragma unroll
    for (int it = 0; it < 4; ++it) {
      int chunk = tid + it * 256;  // A: 128x64 -> 1024 chunks of 8 halves
      int r = chunk >> 3, c = (chunk & 7) << 3;
      *(v8h*)&As[r][c] = *(const v8h*)(A + (size_t)(m0 + r) * KDIM + k0 + c);
      if (k0 + 64 < KDIM)
        __builtin_prefetch(A + (size_t)(m0 + r) * KDIM + k0 + 64 + c, 0, 1);
    }
#pragma unroll
    for (int it = 0; it < 2; ++it) {
      int chunk = tid + it * 256;  // B: 64x64 -> 512 chunks
      int r = chunk >> 3, c = (chunk & 7) << 3;
      *(v8h*)&Bs[r][c] = *(const v8h*)(Bt + (size_t)(n0 + r) * KDIM + k0 + c);
      if (k0 + 64 < KDIM)
        __builtin_prefetch(Bt + (size_t)(n0 + r) * KDIM + k0 + 64 + c, 0, 1);
    }
    __syncthreads();
    gemm_compute_step((const half_t(*)[72])As, (const half_t(*)[72])Bs, lane,
                      wm, wn, acc);
    __syncthreads();
  }
#endif
}

// ---------------------------------------------------------------- conversions
__global__ void k_f32_to_f16(const float* __restrict__ in,
                             half_t* __restrict__ out, int n) {
  int i = blockIdx.x * blockDim.x + threadIdx.x;
  int stride = gridDim.x * blockDim.x;
  for (; i < n; i += stride) out[i] = (half_t)in[i];
}

// out[n][k] = (f16) in[k][n]   for 1024x1024
__global__ void k_transpose_f16(const float* __restrict__ in,
                                half_t* __restrict__ out) {
  __shared__ half_t t[32][33];
  const int bx = blockIdx.x * 32;  // n base
  const int by = blockIdx.y * 32;  // k base
  for (int i = threadIdx.y; i < 32; i += 8)
    t[i][threadIdx.x] = (half_t)in[(size_t)(by + i) * DMODEL + bx + threadIdx.x];
  __syncthreads();
  for (int i = threadIdx.y; i < 32; i += 8)
    out[(size_t)(bx + i) * DMODEL + by + threadIdx.x] = t[threadIdx.x][i];
}

// ---------------------------------------------------------------- QKV GEMM
__global__ __launch_bounds__(256) void k_gemm_qkv(
    const half_t* __restrict__ A, const half_t* __restrict__ WT,
    half_t* __restrict__ Qm, half_t* __restrict__ Km, half_t* __restrict__ VTm) {
  __shared__ half_t As[2 * 128][72];
  __shared__ half_t Bs[2 * 64][72];
  const int which = blockIdx.z;
  const half_t* Bt = WT + (size_t)which * KDIM * DMODEL;
  const int m0 = blockIdx.y * 128;
  const int n0 = blockIdx.x * 64;
  const int tid = threadIdx.x;
  const int lane = tid & 31, wid = tid >> 5;
  const int wm = (wid & 3) * 32;
  const int wn = (wid >> 2) * 32;
  const int cl = lane & 15;
  const int hi = lane >> 4;

  v8f acc[2][2];
#pragma unroll
  for (int i = 0; i < 2; ++i)
#pragma unroll
    for (int j = 0; j < 2; ++j) acc[i][j] = vzero();

  gemm_mainloop(A, Bt, MROWS, DMODEL, m0, n0, tid, As, Bs, acc);

  const float scale = (which == 0) ? 0.125f : 1.0f;  // 1/sqrt(HDIM) for Q
#pragma unroll
  for (int mt = 0; mt < 2; ++mt)
#pragma unroll
    for (int nt = 0; nt < 2; ++nt)
#pragma unroll
      for (int r = 0; r < 8; ++r) {
        int row = m0 + wm + mt * 16 + (hi << 3) + r;  // [0,8192)
        int col = n0 + wn + nt * 16 + cl;             // [0,1024)
        int b = row >> 11, i = row & (NSEQ - 1);
        int h = col >> 6, d = col & (HDIM - 1);
        half_t val = (half_t)(acc[mt][nt][r] * scale);
        size_t bh = (size_t)(b * NHEAD + h);
        if (which == 0)      Qm[(bh * NSEQ + i) * HDIM + d] = val;
        else if (which == 1) Km[(bh * NSEQ + i) * HDIM + d] = val;
        else                 VTm[(bh * HDIM + d) * NSEQ + i] = val;
      }
}

// ---------------------------------------------------------------- attention
__global__ __launch_bounds__(128) void k_attn(
    const half_t* __restrict__ Q, const half_t* __restrict__ Kq,
    const half_t* __restrict__ VT, const int* __restrict__ amask,
    half_t* __restrict__ ctx) {
  __shared__ half_t pS[4][16][48];
  const int lane = threadIdx.x & 31;
  const int w = threadIdx.x >> 5;
  const int cl = lane & 15;
  const int hi = lane >> 4;
  const int row_half = hi << 3;

  const int bh = blockIdx.y;
  const int b = bh >> 4;
  const int h = bh & (NHEAD - 1);
  const int q0 = (blockIdx.x * 4 + w) * 16;

  const half_t* Qh = Q + (size_t)bh * NSEQ * HDIM;
  const half_t* Kh = Kq + (size_t)bh * NSEQ * HDIM;
  const half_t* Vh = VT + (size_t)bh * HDIM * NSEQ;
  const int* mk = amask + b * NSEQ;

  v16h aq[2];
  {
    const half_t* qrow = Qh + (size_t)(q0 + cl) * HDIM;
#pragma unroll
    for (int kk = 0; kk < 2; ++kk)
      aq[kk] = make_frag(*(const v8h*)(qrow + kk * 32 + (hi << 3)),
                         *(const v8h*)(qrow + kk * 32 + 16 + (hi << 3)));
  }

  v8f cacc[4];
#pragma unroll
  for (int nt = 0; nt < 4; ++nt) cacc[nt] = vzero();
  float m_r[8], l_r[8];
#pragma unroll
  for (int r = 0; r < 8; ++r) { m_r[r] = -1e30f; l_r[r] = 0.f; }

  for (int j0 = 0; j0 <= q0 + 15; j0 += 32) {
    v8f s[2];
    s[0] = vzero(); s[1] = vzero();
#pragma unroll
    for (int jt = 0; jt < 2; ++jt) {
      const half_t* krow = Kh + (size_t)(j0 + jt * 16 + cl) * HDIM;
#pragma unroll
      for (int kk = 0; kk < 2; ++kk) {
        const half_t* p = krow + kk * 32 + (hi << 3);
        v16h bk = make_frag(*(const v8h*)p, *(const v8h*)(p + 16));
        s[jt] = wmma_f16(aq[kk], bk, s[jt]);
      }
    }
    const int key0 = j0 + cl;
    const int key1 = j0 + 16 + cl;
    const int pad0 = mk[key0];
    const int pad1 = mk[key1];
#pragma unroll
    for (int r = 0; r < 8; ++r) {
      const int qg = q0 + row_half + r;
      float v0 = (pad0 && key0 <= qg) ? s[0][r] : -1e30f;
      float v1 = (pad1 && key1 <= qg) ? s[1][r] : -1e30f;
      float t = fmaxf(v0, v1);
#pragma unroll
      for (int off = 1; off < 16; off <<= 1) t = fmaxf(t, __shfl_xor(t, off, 16));
      const float mnew = fmaxf(m_r[r], t);
      const float alpha = __expf(m_r[r] - mnew);
      const float p0 = __expf(v0 - mnew);
      const float p1 = __expf(v1 - mnew);
      float ps = p0 + p1;
#pragma unroll
      for (int off = 1; off < 16; off <<= 1) ps += __shfl_xor(ps, off, 16);
      l_r[r] = l_r[r] * alpha + ps;
      m_r[r] = mnew;
#pragma unroll
      for (int nt = 0; nt < 4; ++nt) cacc[nt][r] *= alpha;
      s[0][r] = p0;
      s[1][r] = p1;
    }
#pragma unroll
    for (int r = 0; r < 8; ++r) {
      pS[w][row_half + r][cl] = (half_t)s[0][r];
      pS[w][row_half + r][16 + cl] = (half_t)s[1][r];
    }
    asm volatile("" ::: "memory");  // DS store->load same-wave in-order
    const half_t* pp = &pS[w][cl][0];
    v16h pf = make_frag(*(const v8h*)(pp + row_half),
                        *(const v8h*)(pp + row_half + 16));
#pragma unroll
    for (int nt = 0; nt < 4; ++nt) {
      const half_t* vrow = Vh + (size_t)(nt * 16 + cl) * NSEQ + j0 + (hi << 3);
      v16h vf = make_frag(*(const v8h*)vrow, *(const v8h*)(vrow + 16));
      cacc[nt] = wmma_f16(pf, vf, cacc[nt]);
    }
  }

#pragma unroll
  for (int r = 0; r < 8; ++r) {
    const float inv = 1.0f / l_r[r];
    const int i = q0 + row_half + r;
    const size_t rowbase = ((size_t)b * NSEQ + i) * DMODEL + h * HDIM;
#pragma unroll
    for (int nt = 0; nt < 4; ++nt)
      ctx[rowbase + nt * 16 + cl] = (half_t)(cacc[nt][r] * inv);
  }
}

// ---------------------------------------------------------------- out GEMM
__global__ __launch_bounds__(256) void k_gemm_out(
    const half_t* __restrict__ A, const half_t* __restrict__ Bt,
    const float* __restrict__ bias, float* __restrict__ C) {
  __shared__ half_t As[2 * 128][72];
  __shared__ half_t Bs[2 * 64][72];
  const int m0 = blockIdx.y * 128;
  const int n0 = blockIdx.x * 64;
  const int tid = threadIdx.x;
  const int lane = tid & 31, wid = tid >> 5;
  const int wm = (wid & 3) * 32;
  const int wn = (wid >> 2) * 32;
  const int cl = lane & 15;
  const int hi = lane >> 4;

  v8f acc[2][2];
#pragma unroll
  for (int i = 0; i < 2; ++i)
#pragma unroll
    for (int j = 0; j < 2; ++j) acc[i][j] = vzero();

  gemm_mainloop(A, Bt, MROWS, DMODEL, m0, n0, tid, As, Bs, acc);

#pragma unroll
  for (int mt = 0; mt < 2; ++mt)
#pragma unroll
    for (int nt = 0; nt < 2; ++nt)
#pragma unroll
      for (int r = 0; r < 8; ++r) {
        int row = m0 + wm + mt * 16 + (hi << 3) + r;
        int col = n0 + wn + nt * 16 + cl;
        C[(size_t)row * DMODEL + col] = acc[mt][nt][r] + bias[col];
      }
}

// ---------------------------------------------------------------- launch
extern "C" void kernel_launch(void* const* d_in, const int* in_sizes, int n_in,
                              void* d_out, int out_size, void* d_ws,
                              size_t ws_size, hipStream_t stream) {
  const float* x  = (const float*)d_in[0];
  const int*   am = (const int*)d_in[1];
  const float* Wq = (const float*)d_in[2];
  const float* Wk = (const float*)d_in[3];
  const float* Wv = (const float*)d_in[4];
  const float* Wo = (const float*)d_in[5];
  const float* bo = (const float*)d_in[6];
  float* out = (float*)d_out;

  constexpr size_t SZ_XH = (size_t)MROWS * DMODEL;                   // 8M halves
  constexpr size_t SZ_W  = (size_t)DMODEL * DMODEL;                  // 1M halves
  constexpr size_t SZ_HEAD = (size_t)BATCH * NHEAD * NSEQ * HDIM;    // 8M halves
  constexpr size_t NEEDED =
      (SZ_XH + 4 * SZ_W + 3 * SZ_HEAD + SZ_XH) * sizeof(half_t);
  if (ws_size < NEEDED) return;

  half_t* p = (half_t*)d_ws;
  half_t* xh    = p; p += SZ_XH;
  half_t* wtqkv = p; p += 3 * SZ_W;
  half_t* woT   = p; p += SZ_W;
  half_t* Qb    = p; p += SZ_HEAD;
  half_t* Kb    = p; p += SZ_HEAD;
  half_t* VTb   = p; p += SZ_HEAD;
  half_t* ctxh  = p; p += SZ_XH;

  k_f32_to_f16<<<dim3(4096), 256, 0, stream>>>(x, xh, (int)SZ_XH);
  dim3 tb(32, 8);
  k_transpose_f16<<<dim3(32, 32), tb, 0, stream>>>(Wq, wtqkv);
  k_transpose_f16<<<dim3(32, 32), tb, 0, stream>>>(Wk, wtqkv + SZ_W);
  k_transpose_f16<<<dim3(32, 32), tb, 0, stream>>>(Wv, wtqkv + 2 * SZ_W);
  k_transpose_f16<<<dim3(32, 32), tb, 0, stream>>>(Wo, woT);

  k_gemm_qkv<<<dim3(DMODEL / 64, MROWS / 128, 3), 256, 0, stream>>>(
      xh, wtqkv, Qb, Kb, VTb);

  k_attn<<<dim3(NSEQ / 64, BATCH * NHEAD), 128, 0, stream>>>(Qb, Kb, VTb, am,
                                                             ctxh);

  k_gemm_out<<<dim3(DMODEL / 64, MROWS / 128), 256, 0, stream>>>(ctxh, woT, bo,
                                                                 out);
}
